// DGAT_28054726378293
// MI455X (gfx1250) — matrix operand order repaired and checked by reading, not compiled, not run on hardware
//
#include <hip/hip_runtime.h>
#include <hip/hip_bf16.h>
#include <math.h>
#include <stdint.h>

// ---------------------------------------------------------------------------
// DGAT forward for MI455X (gfx1250, wave32, WMMA + TDM).
//   x_in  = elu(GATConv(x, W1, att1))          -> d_out[0      .. N*16)
//   x_out = elu(GATConv(x, W2, att2))          -> d_out[N*16   .. 2*N*16)
//   x_self= elu(elu(x@lin1+b1) @ lin2 + b2)    -> d_out[2*N*16 .. 3*N*16)
//
// GEMM path: x pre-packed to bf16 (row-major), weights pre-packed to bf16 in
// WMMA-B fragment order; per K-step wave0 DMAs the weight slab into LDS via
// tensor_load_to_lds, fragments read as ds_load_b128, v_wmma_f32_16x16x32_bf16
// with f32 accumulate. Edge softmax/aggregation via global float atomics.
// ---------------------------------------------------------------------------

typedef __attribute__((ext_vector_type(16))) __bf16 v16bf;
typedef __attribute__((ext_vector_type(8)))  __bf16 v8bf;
typedef __attribute__((ext_vector_type(8)))  float  v8f;
typedef __attribute__((ext_vector_type(4)))  unsigned int u32x4;
typedef __attribute__((ext_vector_type(8)))  int  i32x8;
typedef __attribute__((ext_vector_type(4)))  int  i32x4;

#define HEADS   32
#define GAT_C   16
#define DIN     256
#define XP_COLS 512   // HEADS * GAT_C
#define MAX_COLTILES 8
#define LDS_HALVES (MAX_COLTILES * 4 * 32 * 16)   // 16384 bf16 = 32 KB slab

#if __has_builtin(__builtin_amdgcn_tensor_load_to_lds)
#define HAVE_TDM 1
#if __has_include(<hip/amd_detail/amd_gfx1250_TDM.h>)
#define TDM_6ARG 1
#endif
#endif

__device__ __forceinline__ float elu1(float v)    { return v > 0.f ? v : (__expf(v) - 1.f); }
__device__ __forceinline__ float lrelu02(float v) { return v > 0.f ? v : 0.2f * v; }

__device__ __forceinline__ void atomic_fmax(float* a, float v) {
    if (v >= 0.f) atomicMax((int*)a, __float_as_int(v));
    else          atomicMin((unsigned int*)a, __float_as_uint(v));
}

#ifdef HAVE_TDM
// 1D TDM copy: `n8` 8-byte elements from global `gsrc` to LDS byte offset `ldsOff`.
// D# per CDNA5 ISA ch.8: group0 = count/lds_addr/global_addr/type, group1 =
// data_size=8B, tensor_dim0 = tile_dim0 = n8, dim0_stride = n8; groups 2/3 NULL-ish.
__device__ __forceinline__ void tdm_load_1d(const void* gsrc, unsigned ldsOff, unsigned n8)
{
    const unsigned long long ga = (unsigned long long)(uintptr_t)gsrc;
    u32x4 g0;
    g0[0] = 1u;                                              // count=1, user mode
    g0[1] = ldsOff;                                          // lds_addr
    g0[2] = (unsigned)(ga & 0xFFFFFFFFu);                    // global_addr[31:0]
    g0[3] = (unsigned)((ga >> 32) & 0x01FFFFFFu) | (2u << 30); // addr[56:32] | type=2
    i32x8 g1;
    g1[0] = (3 << 16);                                       // data_size = 8 bytes
    g1[1] = (int)((n8 & 0xFFFFu) << 16);                     // tensor_dim0[15:0]
    g1[2] = (int)((n8 >> 16) & 0xFFFFu);                     // tensor_dim0[31:16]
    g1[3] = (int)((n8 & 0xFFFFu) << 16);                     // tile_dim0
    g1[4] = 0;                                               // tile_dim1/2 unused
    g1[5] = (int)n8;                                         // tensor_dim0_stride
    g1[6] = 0;
    g1[7] = 0;
    i32x4 z4 = {0, 0, 0, 0};
#ifdef TDM_6ARG
    i32x8 z8 = {0, 0, 0, 0, 0, 0, 0, 0};
    __builtin_amdgcn_tensor_load_to_lds(g0, g1, z4, z4, z8, 0);
#else
    __builtin_amdgcn_tensor_load_to_lds(g0, g1, z4, z4, 0);
#endif
}
#endif

// ---------------------------------------------------------------------------
// pack x (f32 row-major) -> bf16 row-major; thread per 8 elements
// ---------------------------------------------------------------------------
__global__ void pack_x_bf16(const float* __restrict__ x, __bf16* __restrict__ xb,
                            long long count8)
{
    const long long t = blockIdx.x * (long long)blockDim.x + threadIdx.x;
    if (t >= count8) return;
    const float* p = x + t * 8;
    v8bf o;
    #pragma unroll
    for (int i = 0; i < 8; ++i) o[i] = (__bf16)p[i];
    *(v8bf*)(xb + t * 8) = o;
}

// ---------------------------------------------------------------------------
// pack weight B[K,NC] (f32 row-major) into bf16 WMMA-B fragment order:
// fragment f = ks*(colTiles*4) + ct*4 + j; within fragment, lane l holds 16
// contiguous bf16: col = ct*64 + j*16 + (l&15), k = ks*32 + (l>>4)*16 + i.
// One thread per (fragment, lane).
// ---------------------------------------------------------------------------
__global__ void pack_w_bf16(const float* __restrict__ B, __bf16* __restrict__ Bp,
                            int K, int NC)
{
    const int colTiles = NC >> 6;
    const int total = (K >> 5) * colTiles * 4 * 32;
    const int t = blockIdx.x * blockDim.x + threadIdx.x;
    if (t >= total) return;
    const int lane = t & 31;
    const int frag = t >> 5;
    const int j  = frag & 3;
    const int ct = (frag >> 2) % colTiles;
    const int ks = frag / (4 * colTiles);
    const int col   = ct * 64 + j * 16 + (lane & 15);
    const int kbase = ks * 32 + ((lane >> 4) << 4);
    v16bf o;
    #pragma unroll
    for (int i = 0; i < 16; ++i) o[i] = (__bf16)B[(size_t)(kbase + i) * NC + col];
    *(v16bf*)(Bp + (size_t)t * 16) = o;
}

// ---------------------------------------------------------------------------
// C[M,NC] = Xb[M,K] @ W  (bf16 WMMA, f32 accumulate).
// blockIdx.x = rowTile (16 rows); wave w = colTile (64 cols each).
// Per K-step wave0 DMAs the packed-W slab (colTiles*4 fragments) to LDS.
// ---------------------------------------------------------------------------
__global__ void wmma_gemm_bf16(const __bf16* __restrict__ Xb, const __bf16* __restrict__ Wp,
                               float* __restrict__ C, int M, int K, int NC,
                               const float* __restrict__ bias, int act)
{
    __shared__ __align__(32) __bf16 ldsB[LDS_HALVES];

    const int colTiles = NC >> 6;
    const int lane = threadIdx.x & 31;
    const int w    = threadIdx.x >> 5;
    const int r0   = blockIdx.x << 4;
    const int ksteps = K >> 5;
    const size_t slabHalves = (size_t)colTiles * 4 * 32 * 16;   // bf16 per K-step

    const int rowA = min(r0 + (lane & 15), M - 1);
    const __bf16* Arow = Xb + (size_t)rowA * K;
    const int kShiftA = (lane >> 4) << 4;                       // 0 or 16
    const int bcol = lane & 15;
    const int c0 = w << 6;

    v8f acc0 = {}, acc1 = {}, acc2 = {}, acc3 = {};

    for (int ks = 0; ks < ksteps; ++ks) {
        // --- stage packed-W slab for this K-step into LDS ---
#ifdef HAVE_TDM
        if (w == 0) {
            tdm_load_1d(Wp + (size_t)ks * slabHalves, 0u, (unsigned)(slabHalves >> 2));
            __builtin_amdgcn_s_wait_tensorcnt(0);
        }
#else
        {
            const unsigned long long* g =
                (const unsigned long long*)(Wp + (size_t)ks * slabHalves);
            unsigned long long* l = (unsigned long long*)ldsB;
            for (int i = threadIdx.x; i < (int)(slabHalves >> 2); i += blockDim.x)
                l[i] = g[i];
        }
#endif
        __syncthreads();
        // The TDM engine wrote ldsB behind the compiler's back: escape the
        // address and clobber memory so the fragment loads below are real.
        asm volatile("" : : "v"((void*)ldsB) : "memory");

        if (w < colTiles) {
            const int k0 = ks << 5;
            // A fragment: two contiguous 16B chunks per lane
            const v8bf alo = *(const v8bf*)(Arow + k0 + kShiftA);
            const v8bf ahi = *(const v8bf*)(Arow + k0 + kShiftA + 8);
            v16bf a;
            #pragma unroll
            for (int i = 0; i < 8; ++i) { a[i] = alo[i]; a[i + 8] = ahi[i]; }

            // B fragments straight out of LDS (32B aligned, ds_load_b128 x2 each)
            const __bf16* f = ldsB + ((size_t)(w * 4) * 32 + lane) * 16;
            const v16bf b0 = *(const v16bf*)(f);
            const v16bf b1 = *(const v16bf*)(f + 32 * 16);
            const v16bf b2 = *(const v16bf*)(f + 2 * 32 * 16);
            const v16bf b3 = *(const v16bf*)(f + 3 * 32 * 16);

            acc0 = __builtin_amdgcn_wmma_f32_16x16x32_bf16(false, a, false, b0, (short)0, acc0, false, false);
            acc1 = __builtin_amdgcn_wmma_f32_16x16x32_bf16(false, a, false, b1, (short)0, acc1, false, false);
            acc2 = __builtin_amdgcn_wmma_f32_16x16x32_bf16(false, a, false, b2, (short)0, acc2, false, false);
            acc3 = __builtin_amdgcn_wmma_f32_16x16x32_bf16(false, a, false, b3, (short)0, acc3, false, false);
        }
        __syncthreads();   // protect LDS slab before next DMA
    }

    if (w >= colTiles) return;

    // C/D layout: elem i -> row r0 + i + (lane<16 ? 0 : 8), col = c0 + 16*tile + lane%16
    const int rbase = r0 + ((lane >> 4) << 3);
    #pragma unroll
    for (int i = 0; i < 8; ++i) {
        const int row = rbase + i;
        if (row >= M) continue;
        float o0 = acc0[i], o1 = acc1[i], o2 = acc2[i], o3 = acc3[i];
        if (bias) {
            o0 += bias[c0 + bcol];      o1 += bias[c0 + 16 + bcol];
            o2 += bias[c0 + 32 + bcol]; o3 += bias[c0 + 48 + bcol];
        }
        if (act) { o0 = elu1(o0); o1 = elu1(o1); o2 = elu1(o2); o3 = elu1(o3); }
        float* cr = C + (size_t)row * NC + c0 + bcol;
        cr[0] = o0; cr[16] = o1; cr[32] = o2; cr[48] = o3;
    }
}

// a_s[n,h] = sum_c xp[n,h,c]*att_src[h,c];  a_d likewise.  One thread per (n,h).
__global__ void scores_kernel(const float* __restrict__ xp,
                              const float* __restrict__ att_src,
                              const float* __restrict__ att_dst,
                              float* __restrict__ a_s, float* __restrict__ a_d, int n)
{
    const int t = blockIdx.x * blockDim.x + threadIdx.x;
    if (t >= n * HEADS) return;
    const int h = t & (HEADS - 1);
    const int node = t >> 5;
    const float* v = xp + (size_t)node * XP_COLS + h * GAT_C;
    float ss = 0.f, sd = 0.f;
    #pragma unroll
    for (int c = 0; c < GAT_C; ++c) {
        const float xv = v[c];
        ss += xv * att_src[h * GAT_C + c];
        sd += xv * att_dst[h * GAT_C + c];
    }
    a_s[t] = ss;
    a_d[t] = sd;
}

__global__ void fill_kernel(float* __restrict__ p, long long count, float val)
{
    const long long t = blockIdx.x * (long long)blockDim.x + threadIdx.x;
    if (t < count) p[t] = val;
}

__device__ __forceinline__ void edge_pair(const int* __restrict__ ei, int E, long long e,
                                          int& src, int& dst)
{
    if (e < E) { src = ei[e]; dst = ei[(long long)E + e]; }
    else       { src = dst = (int)(e - E); }     // appended self loops
}

__global__ void edge_max_kernel(const int* __restrict__ ei, int E, int n,
                                const float* __restrict__ a_s, const float* __restrict__ a_d,
                                float* __restrict__ m)
{
    const long long t = blockIdx.x * (long long)blockDim.x + threadIdx.x;
    if (t >= (long long)(E + n) * HEADS) return;
    const int h = (int)(t & (HEADS - 1));
    int src, dst; edge_pair(ei, E, t >> 5, src, dst);
    const float v = lrelu02(a_s[(size_t)src * HEADS + h] + a_d[(size_t)dst * HEADS + h]);
    atomic_fmax(&m[(size_t)dst * HEADS + h], v);
}

__global__ void edge_sum_kernel(const int* __restrict__ ei, int E, int n,
                                const float* __restrict__ a_s, const float* __restrict__ a_d,
                                const float* __restrict__ m, float* __restrict__ s)
{
    const long long t = blockIdx.x * (long long)blockDim.x + threadIdx.x;
    if (t >= (long long)(E + n) * HEADS) return;
    const int h = (int)(t & (HEADS - 1));
    int src, dst; edge_pair(ei, E, t >> 5, src, dst);
    const size_t di = (size_t)dst * HEADS + h;
    const float v = lrelu02(a_s[(size_t)src * HEADS + h] + a_d[di]);
    atomicAdd(&s[di], __expf(v - m[di]));
}

__global__ void edge_agg_kernel(const int* __restrict__ ei, int E, int n,
                                const float* __restrict__ a_s, const float* __restrict__ a_d,
                                const float* __restrict__ m, const float* __restrict__ s,
                                const float* __restrict__ xp, float* __restrict__ agg)
{
    const long long t = blockIdx.x * (long long)blockDim.x + threadIdx.x;
    if (t >= (long long)(E + n) * HEADS) return;
    const int h = (int)(t & (HEADS - 1));
    int src, dst; edge_pair(ei, E, t >> 5, src, dst);
    const size_t di = (size_t)dst * HEADS + h;
    const float v = lrelu02(a_s[(size_t)src * HEADS + h] + a_d[di]);
    const float alpha = __expf(v - m[di]) / (s[di] + 1e-16f);
    const float* xs = xp + (size_t)src * XP_COLS + h * GAT_C;
    float* ag       = agg + (size_t)dst * XP_COLS + h * GAT_C;
    #pragma unroll
    for (int c = 0; c < GAT_C; ++c) atomicAdd(&ag[c], alpha * xs[c]);
}

__global__ void finalize_kernel(const float* __restrict__ agg, const float* __restrict__ bias,
                                float* __restrict__ out, int n)
{
    const int t = blockIdx.x * blockDim.x + threadIdx.x;
    if (t >= n * GAT_C) return;
    const int c = t & (GAT_C - 1);
    const int node = t >> 4;
    float acc = 0.f;
    #pragma unroll
    for (int h = 0; h < HEADS; ++h) acc += agg[(size_t)node * XP_COLS + h * GAT_C + c];
    out[t] = elu1(acc * (1.f / HEADS) + bias[c]);
}

__global__ void lin2_kernel(const float* __restrict__ x1, const float* __restrict__ w,
                            const float* __restrict__ b, float* __restrict__ out, int n)
{
    const int t = blockIdx.x * blockDim.x + threadIdx.x;
    if (t >= n * GAT_C) return;
    const int c = t & (GAT_C - 1);
    const int node = t >> 4;
    const float* hrow = x1 + (size_t)node * 64;
    float acc = b[c];
    #pragma unroll
    for (int k = 0; k < 64; ++k) acc += hrow[k] * w[k * GAT_C + c];
    out[t] = elu1(acc);
}

// ---------------------------------------------------------------------------
extern "C" void kernel_launch(void* const* d_in, const int* in_sizes, int n_in,
                              void* d_out, int out_size, void* d_ws, size_t ws_size,
                              hipStream_t stream)
{
    const float* x    = (const float*)d_in[0];
    const int*   ei   = (const int*)  d_in[1];
    const float* W1   = (const float*)d_in[2];
    const float* as1  = (const float*)d_in[3];
    const float* ad1  = (const float*)d_in[4];
    const float* b1   = (const float*)d_in[5];
    const float* W2   = (const float*)d_in[6];
    const float* as2  = (const float*)d_in[7];
    const float* ad2  = (const float*)d_in[8];
    const float* b2   = (const float*)d_in[9];
    const float* l1w  = (const float*)d_in[10];
    const float* l1b  = (const float*)d_in[11];
    const float* l2w  = (const float*)d_in[12];
    const float* l2b  = (const float*)d_in[13];

    const int n = in_sizes[0] / DIN;
    const int E = in_sizes[1] / 2;
    float* out = (float*)d_out;

    // workspace layout (float units; bf16 buffers occupy half-density tail)
    float* ws   = (float*)d_ws;
    float* xp   = ws;                       // n*512 f32
    float* m    = xp  + (size_t)n * 512;    // n*32
    float* s    = m   + (size_t)n * 32;     // n*32
    float* agg  = s   + (size_t)n * 32;     // n*512
    float* a_s  = agg + (size_t)n * 512;    // n*32
    float* a_d  = a_s + (size_t)n * 32;     // n*32
    float* x1h  = a_d + (size_t)n * 32;     // n*64
    __bf16* xb  = (__bf16*)(x1h + (size_t)n * 64);        // n*256 bf16
    __bf16* wpk = xb + (size_t)n * DIN;                   // up to 256*512 bf16

    const int rowTiles = (n + 15) >> 4;
    const long long nh = (long long)n * HEADS;
    const long long eh = (long long)(E + n) * HEADS;
    const unsigned bNH = (unsigned)((nh + 255) / 256);
    const unsigned bEH = (unsigned)((eh + 255) / 256);
    const unsigned bNC = (unsigned)(((long long)n * GAT_C + 255) / 256);

    // pack x -> bf16 once (shared by all three GEMMs)
    {
        const long long c8 = (long long)n * DIN / 8;
        pack_x_bf16<<<(unsigned)((c8 + 255) / 256), 256, 0, stream>>>(x, xb, c8);
    }

    auto run_conv = [&](const float* W, const float* atts, const float* attd,
                        const float* bias, float* outSeg) {
        const int wthreads = (DIN / 32) * (XP_COLS / 64) * 4 * 32;   // pack threads
        pack_w_bf16<<<(wthreads + 255) / 256, 256, 0, stream>>>(W, wpk, DIN, XP_COLS);
        wmma_gemm_bf16<<<rowTiles, 256, 0, stream>>>(xb, wpk, xp, n, DIN, XP_COLS, nullptr, 0);
        scores_kernel<<<bNH, 256, 0, stream>>>(xp, atts, attd, a_s, a_d, n);
        fill_kernel<<<bNH, 256, 0, stream>>>(m, nh, -INFINITY);
        fill_kernel<<<bNH, 256, 0, stream>>>(s, nh, 0.f);
        fill_kernel<<<(unsigned)(((long long)n * 512 + 255) / 256), 256, 0, stream>>>(agg, (long long)n * 512, 0.f);
        edge_max_kernel<<<bEH, 256, 0, stream>>>(ei, E, n, a_s, a_d, m);
        edge_sum_kernel<<<bEH, 256, 0, stream>>>(ei, E, n, a_s, a_d, m, s);
        edge_agg_kernel<<<bEH, 256, 0, stream>>>(ei, E, n, a_s, a_d, m, s, xp, agg);
        finalize_kernel<<<bNC, 256, 0, stream>>>(agg, bias, outSeg, n);
    };

    run_conv(W1, as1, ad1, b1, out);                       // x_in
    run_conv(W2, as2, ad2, b2, out + (size_t)n * GAT_C);   // x_out

    // MLP branch: x1 = elu(x@lin1+b1) via WMMA (NC=64 -> 1 wave/block), then lin2
    {
        const int wthreads = (DIN / 32) * 1 * 4 * 32;
        pack_w_bf16<<<(wthreads + 255) / 256, 256, 0, stream>>>(l1w, wpk, DIN, 64);
        wmma_gemm_bf16<<<rowTiles, 32, 0, stream>>>(xb, wpk, x1h, n, DIN, 64, l1b, 1);
        lin2_kernel<<<bNC, 256, 0, stream>>>(x1h, l2w, l2b, out + (size_t)2 * n * GAT_C, n);
    }
}
